// ConvEventGRUCellFuseDownsample_ConditionalInput_37735582663140
// MI455X (gfx1250) — matrix operand order repaired and checked by reading, not compiled, not run on hardware
//
#include <hip/hip_runtime.h>

// ---------------------------------------------------------------------------
// ConvEventGRU on gfx1250: implicit-GEMM 3x3 conv via v_wmma_f32_16x16x32_bf16
// + Tensor Data Mover (tensor_load_to_lds) for the LDS halo staging.
// ---------------------------------------------------------------------------

typedef __bf16 v16bf __attribute__((ext_vector_type(16)));
typedef float  v8f   __attribute__((ext_vector_type(8)));
typedef unsigned int u32x4 __attribute__((ext_vector_type(4)));
typedef int          i32x4 __attribute__((ext_vector_type(4)));
typedef int          i32x8 __attribute__((ext_vector_type(8)));

#if defined(__has_builtin)
#if __has_builtin(__builtin_amdgcn_tensor_load_to_lds) && \
    __has_builtin(__builtin_amdgcn_s_wait_tensorcnt)
#define HAVE_TDM 1
#endif
#endif
#ifndef HAVE_TDM
#define HAVE_TDM 0
#endif

union Frag { uint4 u[2]; v16bf v; };

__device__ __forceinline__ __bf16 f2bf(float f) {
  union { float f; unsigned u; } a; a.f = f;
  unsigned r = (a.u + 0x7FFFu + ((a.u >> 16) & 1u)) >> 16;   // RNE
  union { unsigned short s; __bf16 b; } o; o.s = (unsigned short)r;
  return o.b;
}

__device__ __forceinline__ float sigmoidf_(float x) {
  return 1.0f / (1.0f + expf(-x));
}

// Problem constants (match reference setup_inputs)
#define TT 10
#define BB 4
#define CC 64
#define HW 4096          // 64*64
#define W64 64
#define OC1 129
#define OC1P 144         // padded to 9 WMMA M-tiles
#define KCH 18           // 9 taps * 2 K-chunks of 32

// ---------------------------------------------------------------------------
// Pack OIHW fp32 weights into WMMA A-fragment order (bf16).
// wp element index = ((mt*18 + kch)*32 + lane)*16 + e
// 16-bit A 16x32 layout: lanes 0-15: K = e<8 ? e : e+8
//                        lanes 16-31: K = e<8 ? e+8 : e+16
// ---------------------------------------------------------------------------
__global__ void pack_weights(const float* __restrict__ w, __bf16* __restrict__ wp,
                             int mtiles, int ocReal) {
  int idx = blockIdx.x * 256 + threadIdx.x;
  int total = mtiles * KCH * 32 * 16;
  if (idx >= total) return;
  int e    = idx & 15;
  int lane = (idx >> 4) & 31;
  int kch  = (idx >> 9) % KCH;
  int mt   = idx / (KCH * 512);
  int kc   = kch & 1, tap = kch >> 1;
  int klocal = (lane < 16) ? ((e < 8) ? e : e + 8)
                           : ((e < 8) ? e + 8 : e + 16);
  int c = kc * 32 + klocal;                 // input channel 0..63
  int m = mt * 16 + (lane & 15);            // output channel
  float v = 0.f;
  if (m < ocReal)
    v = w[(size_t)((m * CC + c) * 3 + tap / 3) * 3 + (tap % 3)];
  wp[idx] = f2bf(v);
}

__global__ void zero4(uint4* __restrict__ p, int n) {
  int i = blockIdx.x * 256 + threadIdx.x;
  if (i < n) p[i] = make_uint4(0u, 0u, 0u, 0u);
}

// ---------------------------------------------------------------------------
// 3x3 SAME conv as implicit GEMM. Workgroup = 4-row x 64-col stripe (256 px)
// of one image. LDS: 6x66 halo, NHWC bf16, pixel stride padded to 72 halves
// (144B) for 16B-aligned ds_load_b128. Interior rows staged by the Tensor
// Data Mover (per-row D#, pad_enable inserts the 16B/pixel LDS pad); borders
// zeroed by VALU. 8 waves; each wave owns two N-tiles per M-tile so one A
// fragment feeds two v_wmma_f32_16x16x32_bf16.
// ---------------------------------------------------------------------------
__global__ __launch_bounds__(256)
void conv3x3_wmma(const uint4* __restrict__ wpack, const uint4* __restrict__ in_nhwc,
                  const float* __restrict__ bias, float* __restrict__ out,
                  int mtiles, int ocReal, int ocDim) {
  __shared__ uint4 smem4[6 * 66 * 9];       // 57,024 bytes
  const int rb  = blockIdx.x;               // row block 0..15 (4 rows each)
  const int b   = blockIdx.y;               // batch 0..3
  const int tid = threadIdx.x;

#if HAVE_TDM
  // ---- TDM: one descriptor per valid halo row (wave 0 only; EXEC ignored,
  //      issue is per-wave). Row = 64 px * 64ch bf16 = 4096 elements; LDS pad
  //      4 DWORDs every 32 DWORDs -> 144B pixel stride.
  if (tid < 32) {
    i32x4 z4 = {0, 0, 0, 0};
#if __clang_major__ >= 23
    i32x8 z8 = {0, 0, 0, 0, 0, 0, 0, 0};
#endif
    #pragma unroll
    for (int lr = 0; lr < 6; ++lr) {
      int y = rb * 4 - 1 + lr;
      if (y >= 0 && y < W64) {
        unsigned long long ga = (unsigned long long)(const char*)in_nhwc +
                                (unsigned long long)(b * HW + y * W64) * 128ull;
        unsigned lds_off = (unsigned)(size_t)&smem4[(lr * 66 + 1) * 9];
        u32x4 g0;
        g0[0] = 1u;                                   // count=1, load descriptor
        g0[1] = lds_off;                              // LDS byte address
        g0[2] = (unsigned)(ga & 0xFFFFFFFFu);         // global addr [95:64]
        g0[3] = ((unsigned)(ga >> 32) & 0x01FFFFFFu)  // global addr [120:96]
                | (2u << 30);                         // type = 2 (image)
        i32x8 g1;
        g1[0] = (1 << 16)     // data_size = 2 bytes
              | (1 << 20)     // pad_enable
              | (4 << 22)     // pad_interval: 2^(4+1)=32 DWORDs (128B pixel)
              | (3 << 25);    // pad_amount: 4 DWORDs (16B)
        g1[1] = (int)(4096u << 16);   // tensor_dim0 = 4096 (low16 in [31:16])
        g1[2] = (int)(1u << 16);      // tensor_dim0 hi=0 | tensor_dim1 = 1
        g1[3] = (int)(4096u << 16);   // tensor_dim1 hi=0 | tile_dim0 = 4096
        g1[4] = 1;                    // tile_dim1 = 1, tile_dim2 = 0
        g1[5] = 4096;                 // tensor_dim0_stride low32
        g1[6] = (int)(4096u << 16);   // stride0 hi=0 | tensor_dim1_stride lo16
        g1[7] = 0;                    // tensor_dim1_stride hi
#if __clang_major__ >= 23
        __builtin_amdgcn_tensor_load_to_lds(g0, g1, z4, z4, z8, 0);
#else
        __builtin_amdgcn_tensor_load_to_lds(g0, g1, z4, z4, 0);
#endif
      }
    }
  }
  // ---- zero the halo borders / out-of-range rows (cols 0 & 65, edge rows)
  for (int i = tid; i < 6 * 66 * 8; i += 256) {
    int part = i & 7;
    int pix  = i >> 3;
    int lr = pix / 66, lc = pix - lr * 66;
    int y = rb * 4 - 1 + lr;
    int x = lc - 1;
    if (x < 0 || x >= W64 || y < 0 || y >= W64)
      smem4[pix * 9 + part] = make_uint4(0u, 0u, 0u, 0u);
  }
  __syncthreads();                     // all border ds_stores done
  if (tid < 32) __builtin_amdgcn_s_wait_tensorcnt(0);
  __syncthreads();                     // TDM data visible to all waves
#else
  // ---- fallback: manual staging through VGPRs
  for (int i = tid; i < 6 * 66 * 8; i += 256) {
    int part = i & 7;
    int pix  = i >> 3;
    int lr = pix / 66, lc = pix - lr * 66;
    int y = rb * 4 - 1 + lr;
    int x = lc - 1;
    uint4 v = make_uint4(0u, 0u, 0u, 0u);
    if (y >= 0 && y < W64 && x >= 0 && x < W64)
      v = in_nhwc[(size_t)(b * HW + y * W64 + x) * 8 + part];
    smem4[pix * 9 + part] = v;
  }
  __syncthreads();
#endif

  const int lane = tid & 31;
  const int wave = tid >> 5;
  const int hi   = lane >> 4;               // half-wave select
  const int nloc = lane & 15;               // column within N tile

  // two N-tiles per wave per M-tile: one A fragment feeds two WMMAs
  const int np0 = wave * 32 + nloc;         // pixel of N-tile 2*wave
  const int np1 = np0 + 16;                 // pixel of N-tile 2*wave+1
  const int rr  = np0 >> 6;                 // both tiles share the row
  const int cc0 = np0 & 63;
  const int cc1 = np1 & 63;

  for (int mt = 0; mt < mtiles; ++mt) {     // wave-uniform: EXEC stays full
    v8f acc0 = {0.f, 0.f, 0.f, 0.f, 0.f, 0.f, 0.f, 0.f};
    v8f acc1 = {0.f, 0.f, 0.f, 0.f, 0.f, 0.f, 0.f, 0.f};
    const uint4* wbase = wpack + ((size_t)(mt * KCH) * 32 << 1);

    #pragma unroll
    for (int kch = 0; kch < KCH; ++kch) {
      const int tap = kch >> 1, kc = kch & 1;
      const int dy = tap / 3 - 1, dx = tap % 3 - 1;

      Frag A;                                // A fragment: one 32B global load
      const uint4* ap = wbase + (((size_t)kch * 32 + lane) << 1);
      A.u[0] = ap[0];
      A.u[1] = ap[1];

      const int rowBase = (rr + 1 + dy) * 66 + 1 + dx;
      Frag B0, B1;
      const uint4* bp0 = &smem4[(rowBase + cc0) * 9 + kc * 4 + hi];
      B0.u[0] = bp0[0];
      B0.u[1] = bp0[2];
      const uint4* bp1 = &smem4[(rowBase + cc1) * 9 + kc * 4 + hi];
      B1.u[0] = bp1[0];
      B1.u[1] = bp1[2];

      acc0 = __builtin_amdgcn_wmma_f32_16x16x32_bf16(
               false, A.v, false, B0.v, (short)0, acc0, false, false);
      acc1 = __builtin_amdgcn_wmma_f32_16x16x32_bf16(
               false, A.v, false, B1.v, (short)0, acc1, false, false);
    }

    // epilogue: D layout VGPR v -> M = v + hi*8, N = nloc ; fuse bias, NCHW out
    const int pg0 = rb * 256 + np0;
    const int pg1 = rb * 256 + np1;
    #pragma unroll
    for (int v = 0; v < 8; ++v) {
      int oc = mt * 16 + v + hi * 8;
      float bb = (oc < ocReal) ? bias[oc] : 0.f;
      size_t base = (size_t)(b * ocDim + oc) << 12;
      out[base + pg0] = acc0[v] + bb;
      out[base + pg1] = acc1[v] + bb;
    }
  }
}

// ---------------------------------------------------------------------------
// Gates: update/reset sigmoids, threshold, g = prev_y * reset  (NHWC bf16)
// ---------------------------------------------------------------------------
__global__ __launch_bounds__(256)
void gates_kernel(const float* __restrict__ rz, const float* __restrict__ xt,
                  const float* __restrict__ ysAll, float* __restrict__ u_ws,
                  float* __restrict__ thr_ws, __bf16* __restrict__ g_bf, int t) {
  int idx = blockIdx.x * 256 + threadIdx.x;   // 2^20 threads exactly
  int p = idx & (HW - 1);
  int c = (idx >> 12) & (CC - 1);
  int b = idx >> 18;

  float z  = rz[((size_t)(b * OC1P + c)) * HW + p];
  float r  = rz[((size_t)(b * OC1P + CC + c)) * HW + p];
  float xz = xt[((size_t)((t * BB + b) * (3 * CC) + c)) * HW + p];
  float xr = xt[((size_t)((t * BB + b) * (3 * CC) + CC + c)) * HW + p];

  float u  = sigmoidf_(z + xz);
  float rg = sigmoidf_(r + xr);

  float py = 0.f;
  if (t > 0)
    py = ysAll[((size_t)(((t - 1) * BB + b) * CC + c)) * HW + p];

  u_ws[((size_t)(b * CC + c)) * HW + p] = u;
  g_bf[((size_t)(b * HW + p)) * CC + c] = f2bf(py * rg);

  if (c == 0) {
    float tl = rz[((size_t)(b * OC1P + 2 * CC)) * HW + p];
    thr_ws[b * HW + p] = sigmoidf_(tl);
  }
}

// ---------------------------------------------------------------------------
// Update: tanh input gate, GRU blend, spike threshold, outputs + next state
// ---------------------------------------------------------------------------
__global__ __launch_bounds__(256)
void update_kernel(const float* __restrict__ fc, const float* __restrict__ xt,
                   const float* __restrict__ u_ws, const float* __restrict__ thr_ws,
                   float* __restrict__ h_ws, float* __restrict__ outp,
                   __bf16* __restrict__ prevy_bf, int t) {
  int idx = blockIdx.x * 256 + threadIdx.x;
  int p = idx & (HW - 1);
  int c = (idx >> 12) & (CC - 1);
  int b = idx >> 18;

  size_t sidx = ((size_t)(b * CC + c)) * HW + p;
  float f  = fc[sidx] + xt[((size_t)((t * BB + b) * (3 * CC) + 2 * CC + c)) * HW + p];
  float ig = tanhf(f);
  float u  = u_ws[sidx];
  float hp = (t > 0) ? h_ws[sidx] : 0.f;
  float h  = (1.f - u) * hp + u * ig;

  float thr = thr_ws[b * HW + p];
  float d   = h - thr;
  float ev  = (d > 0.f) ? 1.f : 0.f;
  float y   = h * ev;
  float nd  = (d < 0.f) ? (thr - h) : 0.f;

  const size_t E = (size_t)TT * BB * CC * HW;
  size_t oidx = ((size_t)((t * BB + b) * CC + c)) * HW + p;
  outp[oidx]         = y;
  outp[E + oidx]     = ev;
  outp[2 * E + oidx] = nd;

  h_ws[sidx] = h - ev * thr;
  prevy_bf[((size_t)(b * HW + p)) * CC + c] = f2bf(y);
}

// ---------------------------------------------------------------------------
extern "C" void kernel_launch(void* const* d_in, const int* in_sizes, int n_in,
                              void* d_out, int out_size, void* d_ws, size_t ws_size,
                              hipStream_t stream) {
  (void)in_sizes; (void)n_in; (void)out_size; (void)ws_size;
  const float* xt   = (const float*)d_in[0];
  const float* w_rz = (const float*)d_in[1];
  const float* b_rz = (const float*)d_in[2];
  const float* w_f  = (const float*)d_in[3];
  const float* b_f  = (const float*)d_in[4];
  float* outp = (float*)d_out;

  char* ws = (char*)d_ws;
  size_t off = 0;
  auto alloc = [&](size_t bytes) -> void* {
    void* p = ws + off;
    off = (off + bytes + 255) & ~(size_t)255;
    return p;
  };
  __bf16* wp1     = (__bf16*)alloc((size_t)9 * KCH * 32 * 16 * 2);   // 165,888 B
  __bf16* wp2     = (__bf16*)alloc((size_t)4 * KCH * 32 * 16 * 2);   //  73,728 B
  float*  rz      = (float*) alloc((size_t)BB * OC1P * HW * 4);      // 9.4 MB
  float*  fc      = (float*) alloc((size_t)BB * CC * HW * 4);        // 4 MB
  float*  u_ws    = (float*) alloc((size_t)BB * CC * HW * 4);        // 4 MB
  float*  thr_ws  = (float*) alloc((size_t)BB * HW * 4);             // 64 KB
  float*  h_ws    = (float*) alloc((size_t)BB * CC * HW * 4);        // 4 MB
  __bf16* g_bf    = (__bf16*)alloc((size_t)BB * HW * CC * 2);        // 2 MB
  __bf16* prevy   = (__bf16*)alloc((size_t)BB * HW * CC * 2);        // 2 MB

  // pack weights into WMMA A-fragment order
  pack_weights<<<(9 * KCH * 32 * 16 + 255) / 256, 256, 0, stream>>>(w_rz, wp1, 9, OC1);
  pack_weights<<<(4 * KCH * 32 * 16 + 255) / 256, 256, 0, stream>>>(w_f,  wp2, 4, CC);

  // prev_y (bf16 NHWC) starts at zero
  zero4<<<((BB * HW * CC * 2 / 16) + 255) / 256, 256, 0, stream>>>(
      (uint4*)prevy, BB * HW * CC * 2 / 16);

  dim3 gGemm(16, BB);      // 16 row-stripes x 4 images
  int  nElem = BB * CC * HW;

  for (int t = 0; t < TT; ++t) {
    // rz = conv(prev_y, w_rz) + b_rz      (M=144 padded, K=576)
    conv3x3_wmma<<<gGemm, 256, 0, stream>>>(
        (const uint4*)wp1, (const uint4*)prevy, b_rz, rz, 9, OC1, OC1P);
    // gates + g = prev_y * reset (bf16 NHWC for conv2)
    gates_kernel<<<nElem / 256, 256, 0, stream>>>(
        rz, xt, outp, u_ws, thr_ws, g_bf, t);
    // fc = conv(g, w_f) + b_f
    conv3x3_wmma<<<gGemm, 256, 0, stream>>>(
        (const uint4*)wp2, (const uint4*)g_bf, b_f, fc, 4, CC, CC);
    // GRU blend + spike + outputs + next state
    update_kernel<<<nElem / 256, 256, 0, stream>>>(
        fc, xt, u_ws, thr_ws, h_ws, outp, prevy, t);
  }
}
